// BNN_13271448945337
// MI455X (gfx1250) — matrix operand order repaired and checked by reading, not compiled, not run on hardware
//
#include <hip/hip_runtime.h>
#include <math.h>

// ---------------- vector types ----------------
typedef __attribute__((ext_vector_type(16))) _Float16 v16h;
typedef __attribute__((ext_vector_type(8)))  _Float16 v8h;
typedef __attribute__((ext_vector_type(4)))  _Float16 v4h;
typedef __attribute__((ext_vector_type(8)))  float    v8f;
typedef __attribute__((ext_vector_type(4)))  unsigned int u32x4;
typedef __attribute__((ext_vector_type(8)))  int i32x8;
typedef __attribute__((ext_vector_type(4)))  int i32x4;

// ---------------- problem dims ----------------
#define BROWS   65536
#define D_IN    256
#define HDIM    512
#define D_OUT   128
#define M_TILE  32                 // rows per workgroup
#define PITCH_X (D_IN + 8)         // 264 halves (16B pad -> no LDS bank conflicts)
#define PITCH_H (HDIM + 8)         // 520 halves
#define WPITCH  40                 // 32 K-halves + 8 pad halves per weight row in LDS
#define WBUF_HALVES (HDIM * WPITCH) // 20480 halves = 40KB per buffer

#define GAMMA_SHAPE 1.2f           // 12/sqrt(100)

// ---------------- helpers ----------------
__device__ __forceinline__ unsigned lds_off(const void* p) {
  // generic pointer derived from __shared__: low 32 bits == LDS byte offset
  return (unsigned)(unsigned long long)(uintptr_t)p;
}

// Issue one TDM 2-D tile load: [nrows] rows x 32 f16 K-elements, row stride = kd
// elems, into LDS with HW padding of 4 dwords after every 16 dwords (WPITCH=40).
__device__ __forceinline__ void tdm_load_wtile(const _Float16* gsrc,
                                               unsigned ldsoff,
                                               int kd, int nrows) {
  unsigned long long ga = (unsigned long long)(uintptr_t)gsrc;
  u32x4 g0;
  g0[0] = 1u;                                      // count=1 (valid descriptor)
  g0[1] = ldsoff;                                  // lds_addr (bytes)
  g0[2] = (unsigned)(ga & 0xffffffffu);            // global_addr[31:0]
  g0[3] = (unsigned)((ga >> 32) & 0x01ffffffu)     // global_addr[56:32]
        | 0x80000000u;                             // type=2 ("image")
  i32x8 g1;
  // data_size=1 (2B) @16; pad_enable @20; pad_interval=3 (16 dw) @22;
  // pad_amount=3 (4 dw) @25; workgroup_mask=0
  g1[0] = (int)0x06D10000u;
  g1[1] = (int)(((unsigned)kd & 0xffffu) << 16);   // tensor_dim0[15:0] in hi half
  g1[2] = (int)(((unsigned)nrows & 0xffffu) << 16);// dim0 hi=0 | tensor_dim1 lo
  g1[3] = (int)(32u << 16);                        // dim1 hi=0 | tile_dim0=32
  g1[4] = nrows;                                   // tile_dim1=nrows, tile_dim2=0
  g1[5] = kd;                                      // tensor_dim0_stride[31:0]
  g1[6] = 0;                                       // stride hi | dim1_stride lo
  g1[7] = 0;
  i32x4 z4 = {0, 0, 0, 0};
#if __clang_major__ >= 23
  i32x8 z8 = {0, 0, 0, 0, 0, 0, 0, 0};
  __builtin_amdgcn_tensor_load_to_lds(g0, g1, z4, z4, z8, 0);
#else
  __builtin_amdgcn_tensor_load_to_lds(g0, g1, z4, z4, 0);
#endif
}

// ---------------- weight f32 -> f16 transpose pre-pass ----------------
// ws layout (halves): Wt1[512][256] @0 | Wt2[512][512] @131072 | Wt3[128][512] @393216
__global__ void convert_weights(const float* __restrict__ W1,
                                const float* __restrict__ W2,
                                const float* __restrict__ W3,
                                _Float16* __restrict__ ws) {
  const int n1 = HDIM * D_IN;     // 131072
  const int n2 = HDIM * HDIM;     // 262144
  const int n3 = D_OUT * HDIM;    // 65536
  int i = blockIdx.x * blockDim.x + threadIdx.x;
  if (i < n1) {
    int n = i / D_IN, k = i % D_IN;
    ws[i] = (_Float16)W1[k * HDIM + n];
  } else if (i < n1 + n2) {
    int j = i - n1; int n = j / HDIM, k = j % HDIM;
    ws[n1 + j] = (_Float16)W2[k * HDIM + n];
  } else if (i < n1 + n2 + n3) {
    int j = i - n1 - n2; int n = j / HDIM, k = j % HDIM;
    ws[n1 + n2 + j] = (_Float16)W3[k * D_OUT + n];
  }
}

// ---------------- one MLP layer on a 32-row tile ----------------
// sIn: [M_TILE][KD] f16 (pitch pitchIn halves). wt: [ND][KD] f16 row-major in L2.
// Output: tanh(acc + bias) -> sOut f16 (pitch pitchOut) or gout f32 [.,ND].
template <int KD, int ND, bool TOGLOBAL>
__device__ __forceinline__ void layer_run(const _Float16* __restrict__ sIn,
                                          int pitchIn,
                                          const _Float16* __restrict__ wt,
                                          const float* __restrict__ bias,
                                          _Float16* __restrict__ sOut,
                                          int pitchOut,
                                          float* __restrict__ gout, int rowBase,
                                          _Float16 (*sW)[WBUF_HALVES]) {
  constexpr int NK  = KD / 32;        // WMMA K-steps
  constexpr int NBW = ND / 64;        // N-blocks per wave (8 waves, 2 M-blocks)
  const int tid  = threadIdx.x;
  const int lane = tid & 31;
  const int wv   = tid >> 5;
  const int lm   = lane & 15;         // M/N index within a 16x16 tile
  const int lh   = lane >> 4;         // lane-half select
  const int mbase  = (wv & 1) * 16;   // waves 0,2,4,6 -> rows 0..15; 1,3,5,7 -> 16..31
  const int nbBase = (wv >> 1) * NBW; // 4 groups of N-blocks

  v8f acc[NBW];
#pragma unroll
  for (int i = 0; i < NBW; ++i)
#pragma unroll
    for (int r = 0; r < 8; ++r) acc[i][r] = 0.0f;

  if (wv == 0) tdm_load_wtile(wt, lds_off(&sW[0][0]), KD, ND);   // prologue: kt=0

  for (int kt = 0; kt < NK; ++kt) {
    if (wv == 0) {
      if (kt + 1 < NK) {   // prefetch next K-slab into the other buffer
        tdm_load_wtile(wt + (kt + 1) * 32, lds_off(&sW[(kt + 1) & 1][0]), KD, ND);
        __builtin_amdgcn_s_wait_tensorcnt(1);   // in-order: slab kt is complete
      } else {
        __builtin_amdgcn_s_wait_tensorcnt(0);
      }
    }
    __syncthreads();

    // A fragment (16x32 f16): two contiguous b128 chunks per lane
    const v8h* pa = (const v8h*)(sIn + (size_t)(mbase + lm) * pitchIn +
                                 kt * 32 + 8 * lh);
    v8h a0 = pa[0];
    v8h a1 = pa[2];   // +16 halves
    v16h av = __builtin_shufflevector(a0, a1, 0, 1, 2, 3, 4, 5, 6, 7,
                                      8, 9, 10, 11, 12, 13, 14, 15);

    const _Float16* wbuf = &sW[kt & 1][0];
#pragma unroll
    for (int nb = 0; nb < NBW; ++nb) {
      // B fragment (32x16 f16): lane = column N, K chunks 8*lh and 16+8*lh per lane
      const v8h* pb = (const v8h*)(wbuf +
          (size_t)((nbBase + nb) * 16 + lm) * WPITCH + 8 * lh);
      v8h b0 = pb[0];
      v8h b1 = pb[2];
      v16h bv = __builtin_shufflevector(b0, b1, 0, 1, 2, 3, 4, 5, 6, 7,
                                        8, 9, 10, 11, 12, 13, 14, 15);
      acc[nb] = __builtin_amdgcn_wmma_f32_16x16x32_f16(
          false, av, false, bv, (short)0, acc[nb], false, false);
    }
    __syncthreads();   // everyone done with this slab before it is overwritten
  }

  // epilogue: bias + tanh
#pragma unroll
  for (int nb = 0; nb < NBW; ++nb) {
    const int col = (nbBase + nb) * 16 + lm;
    const float bv_ = bias[col];
#pragma unroll
    for (int r = 0; r < 8; ++r) {
      const int m = mbase + r + 8 * lh;   // C/D layout: VGPR r -> row r (+8 for hi half)
      float v = tanhf(acc[nb][r] + bv_);
      if constexpr (TOGLOBAL)
        gout[(size_t)(rowBase + m) * ND + col] = v;
      else
        sOut[m * pitchOut + col] = (_Float16)v;
    }
  }
  __syncthreads();
}

// ---------------- fused 3-layer MLP ----------------
__global__ void __launch_bounds__(256, 1)
mlp_fused(const float* __restrict__ x,
          const float* __restrict__ b1, const float* __restrict__ b2,
          const float* __restrict__ b3,
          const _Float16* __restrict__ wt1, const _Float16* __restrict__ wt2,
          const _Float16* __restrict__ wt3,
          float* __restrict__ out) {
  __shared__ __align__(16) _Float16 sR1[M_TILE * PITCH_H]; // x tile, then h2
  __shared__ __align__(16) _Float16 sR2[M_TILE * PITCH_H]; // h1
  __shared__ __align__(16) _Float16 sW[2][WBUF_HALVES];    // double-buffered W slab
  const int tid = threadIdx.x;
  const int rowBase = blockIdx.x * M_TILE;

  // stage x tile: f32 -> f16 into sR1 with pitch PITCH_X
  {
    const float4* xg = (const float4*)(x + (size_t)rowBase * D_IN);
    for (int f = tid; f < M_TILE * (D_IN / 4); f += 256) {
      int row = f >> 6;          // 64 float4 per row
      int k4  = f & 63;
      float4 v = xg[row * 64 + k4];
      v4h h;
      h[0] = (_Float16)v.x; h[1] = (_Float16)v.y;
      h[2] = (_Float16)v.z; h[3] = (_Float16)v.w;
      *(v4h*)(sR1 + (size_t)row * PITCH_X + k4 * 4) = h;
    }
  }
  // (first barrier inside layer_run orders the staging stores)

  layer_run<D_IN, HDIM, false>(sR1, PITCH_X, wt1, b1, sR2, PITCH_H,
                               nullptr, 0, sW);
  layer_run<HDIM, HDIM, false>(sR2, PITCH_H, wt2, b2, sR1, PITCH_H,
                               nullptr, 0, sW);
  layer_run<HDIM, D_OUT, true>(sR1, PITCH_H, wt3, b3, nullptr, 0,
                               out, rowBase, sW);
}

// ---------------- sigma: iid Gamma(1.2, 1) via Marsaglia-Tsang ----------------
__device__ __forceinline__ unsigned pcg_hash(unsigned v) {
  unsigned s = v * 747796405u + 2891336453u;
  unsigned w = ((s >> ((s >> 28u) + 4u)) ^ s) * 277803737u;
  return (w >> 22u) ^ w;
}

__global__ void gamma_kernel(float* __restrict__ out, int n) {
  int i = blockIdx.x * blockDim.x + threadIdx.x;
  if (i >= n) return;
  const float d = GAMMA_SHAPE - (1.0f / 3.0f);
  const float c = 1.0f / sqrtf(9.0f * d);
  unsigned base = (unsigned)i * 2654435761u + 0x9E3779B9u;
  float res = d;   // fallback = mean-ish value
#pragma unroll 1
  for (int it = 0; it < 16; ++it) {
    unsigned r1 = pcg_hash(base + (unsigned)it * 0x85EBCA6Bu);
    unsigned r2 = pcg_hash(base ^ ((unsigned)it * 0xC2B2AE35u + 0x27D4EB2Fu));
    unsigned r3 = pcg_hash(base + 0x165667B1u + (unsigned)it * 0x9E3779B9u);
    float u1 = (float)(r1 >> 8) * (1.0f / 16777216.0f) + 1e-7f;
    float u2 = (float)(r2 >> 8) * (1.0f / 16777216.0f);
    float u  = (float)(r3 >> 8) * (1.0f / 16777216.0f) + 1e-7f;
    float xn = sqrtf(-2.0f * logf(u1)) * cosf(6.28318530718f * u2); // N(0,1)
    float v = 1.0f + c * xn;
    v = v * v * v;
    if (v > 0.0f &&
        logf(u) < 0.5f * xn * xn + d - d * v + d * logf(v)) {
      res = d * v;
      break;
    }
  }
  out[i] = res;
}

// ---------------- launcher ----------------
extern "C" void kernel_launch(void* const* d_in, const int* in_sizes, int n_in,
                              void* d_out, int out_size, void* d_ws,
                              size_t ws_size, hipStream_t stream) {
  const float* x  = (const float*)d_in[0];
  const float* W1 = (const float*)d_in[1];
  const float* b1 = (const float*)d_in[2];
  const float* W2 = (const float*)d_in[3];
  const float* b2 = (const float*)d_in[4];
  const float* W3 = (const float*)d_in[5];
  const float* b3 = (const float*)d_in[6];
  float* out = (float*)d_out;

  _Float16* wt1 = (_Float16*)d_ws;                 // 512*256
  _Float16* wt2 = wt1 + HDIM * D_IN;               // 512*512
  _Float16* wt3 = wt2 + HDIM * HDIM;               // 128*512

  // 1) weights f32 -> f16 transposed into workspace (~900 KB, stays in L2)
  const int wtot = HDIM * D_IN + HDIM * HDIM + D_OUT * HDIM;   // 458752
  convert_weights<<<(wtot + 255) / 256, 256, 0, stream>>>(W1, W2, W3, wt1);

  // 2) fused 3-layer WMMA MLP -> mu
  mlp_fused<<<BROWS / M_TILE, 256, 0, stream>>>(x, b1, b2, b3,
                                                wt1, wt2, wt3, out);

  // 3) sigma samples
  const int nsig = BROWS * D_OUT;                  // 8388608
  gamma_kernel<<<nsig / 256, 256, 0, stream>>>(out + nsig, nsig);
}
